// GAT_22539988369861
// MI455X (gfx1250) — matrix operand order repaired and checked by reading, not compiled, not run on hardware
//
#include <hip/hip_runtime.h>
#include <hip/hip_bf16.h>

// ---------------------------------------------------------------------------
// Problem constants (from reference)
#define NN      50000          // nodes (multiple of 16)
#define EE      800000         // raw edges
#define ETOT    (EE + NN)      // edges + self loops
#define FIN     128
#define HH      128            // HEADS*HID
#define HEADS   4
#define HID     32
#define CC      40
#define CP      48             // padded C for WMMA column tiles
#define K2P     96             // padded K (2C=80 -> 96) for final GEMM
#define NEG_SLOPE 0.2f

static_assert(NN % 16 == 0, "node count must be a multiple of 16");

typedef __attribute__((ext_vector_type(16))) _Float16 v16h;
typedef __attribute__((ext_vector_type(8)))  _Float16 v8h;
typedef __attribute__((ext_vector_type(8)))  float    v8f;

// ------------------------- small helpers -----------------------------------
__device__ __forceinline__ unsigned fenc(float f) {
    unsigned u = __float_as_uint(f);
    return (u & 0x80000000u) ? ~u : (u | 0x80000000u);
}
__device__ __forceinline__ float fdec(unsigned u) {
    u = (u & 0x80000000u) ? (u & 0x7FFFFFFFu) : ~u;
    return __uint_as_float(u);
}
__device__ __forceinline__ float lrelu(float x) { return x > 0.f ? x : NEG_SLOPE * x; }

// ------------------------- workspace zero-fill -----------------------------
__global__ void zero_kernel(float* __restrict__ p, size_t nwords) {
    size_t i = (size_t)blockIdx.x * blockDim.x + threadIdx.x;
    size_t stride = (size_t)gridDim.x * blockDim.x;
    for (; i < nwords; i += stride) p[i] = 0.0f;
}

// ------------------------- weight conversion to f16 (row-major, padded) ----
// W1c   [128][160] : cols 0..127 = gat_W1, cols 128..159 = gcn_W1
// W2p   [128][48]  : gat_W2 padded
// gW2p  [32][48]   : gcn_W2 padded
// linWp [96][48]   : lin_W  padded (K 80->96, N 40->48)
__global__ void conv_weights_kernel(const float* __restrict__ gat_W1,
                                    const float* __restrict__ gcn_W1,
                                    const float* __restrict__ gat_W2,
                                    const float* __restrict__ gcn_W2,
                                    const float* __restrict__ lin_W,
                                    _Float16* __restrict__ W1c,
                                    _Float16* __restrict__ W2p,
                                    _Float16* __restrict__ gW2p,
                                    _Float16* __restrict__ linWp) {
    int t = blockIdx.x * blockDim.x + threadIdx.x;
    if (t < 128 * 160) {
        int k = t / 160, j = t % 160;
        float v = (j < HH) ? gat_W1[k * HH + j] : gcn_W1[k * HID + (j - HH)];
        W1c[t] = (_Float16)v;
    } else if (t < 128 * 160 + 128 * CP) {
        int u = t - 128 * 160;
        int k = u / CP, j = u % CP;
        W2p[u] = (_Float16)((j < CC) ? gat_W2[k * CC + j] : 0.f);
    } else if (t < 128 * 160 + 128 * CP + 32 * CP) {
        int u = t - 128 * 160 - 128 * CP;
        int k = u / CP, j = u % CP;
        gW2p[u] = (_Float16)((j < CC) ? gcn_W2[k * CC + j] : 0.f);
    } else if (t < 128 * 160 + 128 * CP + 32 * CP + K2P * CP) {
        int u = t - 128 * 160 - 128 * CP - 32 * CP;
        int k = u / CP, j = u % CP;
        linWp[u] = (_Float16)((k < 2 * CC && j < CC) ? lin_W[k * CC + j] : 0.f);
    }
}

// ------------------------- pack B into WMMA fragment layout ----------------
// Bsrc row-major [K][ldb] f16 -> Bpk[((tile*(K/32)+kb)*32+lane)*16 + i]
// where for lane: n = lane&15, koff = (lane>>4)*16, element i covers
// K index kb*32+koff+i, column tile*16+n (contiguous k order, same as the A fragment).
__global__ void pack_b_kernel(const _Float16* __restrict__ Bsrc, int ldb,
                              int K, int nct, _Float16* __restrict__ Bpk) {
    int idx = blockIdx.x * blockDim.x + threadIdx.x;
    int total = nct * (K / 32) * 32 * 16;
    if (idx >= total) return;
    int i    = idx & 15;
    int lane = (idx >> 4) & 31;
    int rest = idx >> 9;
    int kb   = rest % (K / 32);
    int tile = rest / (K / 32);
    int n    = lane & 15;
    int koff = (lane >> 4) * 16;
    Bpk[idx] = Bsrc[(size_t)(kb * 32 + koff + i) * ldb + tile * 16 + n];
}

// ------------------------- templated WMMA GEMM -----------------------------
// D[NN x (NCT*16)] = A[NN x K](f16,row-major) * Bpk (fragment-packed).
// Block = 128 threads = 4 waves, one 16-row node tile per block; waves
// stripe column tiles.  K-loop fully unrolled (compile-time K).
template <int LDA, int K, int NCT, int LDD, int NCV, bool BIAS>
__global__ __launch_bounds__(128)
void gemm16_kernel(const _Float16* __restrict__ A,
                   const _Float16* __restrict__ Bpk,
                   float* __restrict__ D,
                   const float* __restrict__ bias) {
    __shared__ __align__(16) _Float16 sA[16 * LDA];
    const int row0 = blockIdx.x * 16;
    constexpr int KB = K / 32;
#pragma unroll
    for (int i = threadIdx.x; i < 16 * LDA; i += 128) {
        int r = i / LDA, c = i % LDA;
        sA[i] = A[(size_t)(row0 + r) * LDA + c];
    }
    __syncthreads();

    const int wave = threadIdx.x >> 5;
    const int lane = threadIdx.x & 31;
    const int m    = lane & 15;           // row (A) / col-within-tile (B,D)
    const int hiA  = (lane >> 4) * 16;    // 16-bit A K-pattern (contiguous, matches Bpk)

    for (int t = wave; t < NCT; t += 4) {
        v8f acc = {};
#pragma unroll
        for (int kb = 0; kb < KB; ++kb) {
            // A fragment from LDS: K = kb*32+hiA..+15
            v8h alo = *(const v8h*)(sA + m * LDA + kb * 32 + hiA);
            v8h ahi = *(const v8h*)(sA + m * LDA + kb * 32 + hiA + 8);
            v16h a;
#pragma unroll
            for (int i = 0; i < 8; ++i) { a[i] = alo[i]; a[8 + i] = ahi[i]; }
            // whole B fragment with one 32B load
            v16h b = *(const v16h*)(Bpk + ((size_t)(t * KB + kb) * 32 + lane) * 16);
            acc = __builtin_amdgcn_wmma_f32_16x16x32_f16(
                false, a, false, b, (short)0, acc, false, false);
        }
        const int col = t * 16 + m;
        float* dp = D + (size_t)(row0 + (lane >> 4) * 8) * LDD + col;
        float bv = 0.f;
        if constexpr (BIAS) { if (col < NCV) bv = bias[col]; }
        if constexpr (NCV % 16 == 0) {
#pragma unroll
            for (int j = 0; j < 8; ++j) dp[(size_t)j * LDD] = acc[j] + bv;
        } else {
            if (col < NCV) {
#pragma unroll
                for (int j = 0; j < 8; ++j) dp[(size_t)j * LDD] = acc[j] + bv;
            }
        }
    }
}

// ------------------------- per-node attention logits -----------------------
__global__ void attn1_kernel(const float* __restrict__ HG1,
                             const float* __restrict__ att_s,
                             const float* __restrict__ att_d,
                             float* __restrict__ as1, float* __restrict__ ad1) {
    int t = blockIdx.x * blockDim.x + threadIdx.x;   // n*4 + h
    if (t >= NN * HEADS) return;
    int n = t >> 2, h = t & 3;
    const float* row = HG1 + (size_t)n * 160 + h * HID;
    float s = 0.f, d = 0.f;
#pragma unroll 8
    for (int k = 0; k < HID; ++k) {
        float v = row[k];
        s += v * att_s[h * HID + k];
        d += v * att_d[h * HID + k];
    }
    as1[t] = s; ad1[t] = d;
}

__global__ void attn2_kernel(const float* __restrict__ H2,
                             const float* __restrict__ att_s,
                             const float* __restrict__ att_d,
                             float* __restrict__ as2, float* __restrict__ ad2) {
    int n = blockIdx.x * blockDim.x + threadIdx.x;
    if (n >= NN) return;
    const float* row = H2 + (size_t)n * CP;
    float s = 0.f, d = 0.f;
#pragma unroll 8
    for (int k = 0; k < CC; ++k) {
        float v = row[k];
        s += v * att_s[k];
        d += v * att_d[k];
    }
    as2[n] = s; ad2[n] = d;
}

// ------------------------- degree / norm -----------------------------------
__global__ void edge_deg_kernel(const int* __restrict__ ei, float* __restrict__ deg) {
    int e = blockIdx.x * blockDim.x + threadIdx.x;
    if (e >= ETOT) return;
    int d = (e < EE) ? ei[EE + e] : (e - EE);
    atomicAdd(&deg[d], 1.0f);
}
__global__ void norm_kernel(const float* __restrict__ deg, float* __restrict__ nrm) {
    int n = blockIdx.x * blockDim.x + threadIdx.x;
    if (n < NN) nrm[n] = rsqrtf(deg[n]);
}

// ------------------------- GAT layer 1 softmax passes ----------------------
__global__ void edge_max1_kernel(const int* __restrict__ ei,
                                 const float* __restrict__ as1,
                                 const float* __restrict__ ad1,
                                 unsigned* __restrict__ m1) {
    int e = blockIdx.x * blockDim.x + threadIdx.x;
    if (e >= ETOT) return;
    int s = (e < EE) ? ei[e] : (e - EE);
    int d = (e < EE) ? ei[EE + e] : (e - EE);
#pragma unroll
    for (int h = 0; h < HEADS; ++h) {
        float x = lrelu(as1[s * HEADS + h] + ad1[d * HEADS + h]);
        atomicMax(&m1[d * HEADS + h], fenc(x));
    }
}
__global__ void edge_sum1_kernel(const int* __restrict__ ei,
                                 const float* __restrict__ as1,
                                 const float* __restrict__ ad1,
                                 const unsigned* __restrict__ m1,
                                 float* __restrict__ s1) {
    int e = blockIdx.x * blockDim.x + threadIdx.x;
    if (e >= ETOT) return;
    int s = (e < EE) ? ei[e] : (e - EE);
    int d = (e < EE) ? ei[EE + e] : (e - EE);
#pragma unroll
    for (int h = 0; h < HEADS; ++h) {
        float x = lrelu(as1[s * HEADS + h] + ad1[d * HEADS + h]);
        atomicAdd(&s1[d * HEADS + h], expf(x - fdec(m1[d * HEADS + h])));
    }
}
// wave-per-edge, lane-per-channel aggregation (GAT 128ch + GCN 32ch fused)
__global__ void edge_agg1_kernel(const int* __restrict__ ei,
                                 const float* __restrict__ as1,
                                 const float* __restrict__ ad1,
                                 const unsigned* __restrict__ m1,
                                 const float* __restrict__ s1,
                                 const float* __restrict__ HG1,
                                 const float* __restrict__ nrm,
                                 float* __restrict__ out1,
                                 float* __restrict__ aggG1) {
    int gt = blockIdx.x * blockDim.x + threadIdx.x;
    int e = gt >> 5, lane = gt & 31;
    if (e >= ETOT) return;
    int s = (e < EE) ? ei[e] : (e - EE);
    int d = (e < EE) ? ei[EE + e] : (e - EE);
    float alpha[HEADS];
#pragma unroll
    for (int h = 0; h < HEADS; ++h) {
        float x = lrelu(as1[s * HEADS + h] + ad1[d * HEADS + h]);
        alpha[h] = expf(x - fdec(m1[d * HEADS + h])) / s1[d * HEADS + h];
    }
#pragma unroll
    for (int i = 0; i < 4; ++i) {
        int c = lane + i * 32;
        atomicAdd(&out1[(size_t)d * HH + c], HG1[(size_t)s * 160 + c] * alpha[i]);
    }
    float cf = nrm[s] * nrm[d];
    atomicAdd(&aggG1[(size_t)d * HID + lane], HG1[(size_t)s * 160 + HH + lane] * cf);
}

// ------------------------- layer-1 activations -> f16 ----------------------
__global__ void act1_kernel(const float* __restrict__ out1,
                            const float* __restrict__ aggG1,
                            const float* __restrict__ gat_b1,
                            const float* __restrict__ gcn_b1,
                            _Float16* __restrict__ xgatH,
                            _Float16* __restrict__ xgcnH) {
    size_t t = (size_t)blockIdx.x * blockDim.x + threadIdx.x;
    if (t >= (size_t)NN * 160) return;
    int n = (int)(t / 160), c = (int)(t % 160);
    if (c < HH) {
        float v = out1[(size_t)n * HH + c] + gat_b1[c];
        v = (v > 0.f) ? v : (expf(v) - 1.0f);               // ELU
        xgatH[(size_t)n * HH + c] = (_Float16)v;
    } else {
        int c2 = c - HH;
        float v = aggG1[(size_t)n * HID + c2] + gcn_b1[c2];
        v = v > 0.f ? v : 0.f;                               // ReLU
        xgcnH[(size_t)n * HID + c2] = (_Float16)v;
    }
}

// ------------------------- GAT layer 2 softmax passes ----------------------
__global__ void edge_max2_kernel(const int* __restrict__ ei,
                                 const float* __restrict__ as2,
                                 const float* __restrict__ ad2,
                                 unsigned* __restrict__ m2) {
    int e = blockIdx.x * blockDim.x + threadIdx.x;
    if (e >= ETOT) return;
    int s = (e < EE) ? ei[e] : (e - EE);
    int d = (e < EE) ? ei[EE + e] : (e - EE);
    atomicMax(&m2[d], fenc(lrelu(as2[s] + ad2[d])));
}
__global__ void edge_sum2_kernel(const int* __restrict__ ei,
                                 const float* __restrict__ as2,
                                 const float* __restrict__ ad2,
                                 const unsigned* __restrict__ m2,
                                 float* __restrict__ s2) {
    int e = blockIdx.x * blockDim.x + threadIdx.x;
    if (e >= ETOT) return;
    int s = (e < EE) ? ei[e] : (e - EE);
    int d = (e < EE) ? ei[EE + e] : (e - EE);
    float x = lrelu(as2[s] + ad2[d]);
    atomicAdd(&s2[d], expf(x - fdec(m2[d])));
}
__global__ void edge_agg2_kernel(const int* __restrict__ ei,
                                 const float* __restrict__ as2,
                                 const float* __restrict__ ad2,
                                 const unsigned* __restrict__ m2,
                                 const float* __restrict__ s2,
                                 const float* __restrict__ H2,
                                 const float* __restrict__ G2,
                                 const float* __restrict__ nrm,
                                 float* __restrict__ aggA2,
                                 float* __restrict__ aggG2) {
    int gt = blockIdx.x * blockDim.x + threadIdx.x;
    int e = gt >> 5, lane = gt & 31;
    if (e >= ETOT) return;
    int s = (e < EE) ? ei[e] : (e - EE);
    int d = (e < EE) ? ei[EE + e] : (e - EE);
    float x = lrelu(as2[s] + ad2[d]);
    float alpha = expf(x - fdec(m2[d])) / s2[d];
    float cf = nrm[s] * nrm[d];
    {
        int c = lane;
        atomicAdd(&aggA2[(size_t)d * CC + c], H2[(size_t)s * CP + c] * alpha);
        atomicAdd(&aggG2[(size_t)d * CC + c], G2[(size_t)s * CP + c] * cf);
    }
    if (lane < CC - 32) {
        int c = lane + 32;
        atomicAdd(&aggA2[(size_t)d * CC + c], H2[(size_t)s * CP + c] * alpha);
        atomicAdd(&aggG2[(size_t)d * CC + c], G2[(size_t)s * CP + c] * cf);
    }
}

// ------------------------- weighted concat -> f16 (K padded to 96) ---------
__global__ void cat_kernel(const float* __restrict__ aggA2,
                           const float* __restrict__ aggG2,
                           const float* __restrict__ gat_b2,
                           const float* __restrict__ gcn_b2,
                           const float* __restrict__ wc_p,
                           const float* __restrict__ wt_p,
                           _Float16* __restrict__ catH) {
    size_t t = (size_t)blockIdx.x * blockDim.x + threadIdx.x;
    if (t >= (size_t)NN * K2P) return;
    int n = (int)(t / K2P), c = (int)(t % K2P);
    float v = 0.f;
    if (c < CC) {
        v = (aggG2[(size_t)n * CC + c] + gcn_b2[c]) * (*wc_p);
    } else if (c < 2 * CC) {
        int c2 = c - CC;
        v = (aggA2[(size_t)n * CC + c2] + gat_b2[c2]) * (*wt_p);
    }
    catH[t] = (_Float16)v;
}

__global__ void f32_to_f16_kernel(const float* __restrict__ src,
                                  _Float16* __restrict__ dst, size_t n) {
    size_t t = (size_t)blockIdx.x * blockDim.x + threadIdx.x;
    if (t < n) dst[t] = (_Float16)src[t];
}

// ---------------------------------------------------------------------------
extern "C" void kernel_launch(void* const* d_in, const int* in_sizes, int n_in,
                              void* d_out, int out_size, void* d_ws, size_t ws_size,
                              hipStream_t stream) {
    const float* x       = (const float*)d_in[0];
    const int*   ei      = (const int*)d_in[1];
    const float* gat_W1  = (const float*)d_in[2];
    const float* att_s1  = (const float*)d_in[3];
    const float* att_d1  = (const float*)d_in[4];
    const float* gat_b1  = (const float*)d_in[5];
    const float* gat_W2  = (const float*)d_in[6];
    const float* att_s2  = (const float*)d_in[7];
    const float* att_d2  = (const float*)d_in[8];
    const float* gat_b2  = (const float*)d_in[9];
    const float* gcn_W1  = (const float*)d_in[10];
    const float* gcn_b1  = (const float*)d_in[11];
    const float* gcn_W2  = (const float*)d_in[12];
    const float* gcn_b2  = (const float*)d_in[13];
    const float* lin_W   = (const float*)d_in[14];
    const float* lin_b   = (const float*)d_in[15];
    const float* wc_p    = (const float*)d_in[16];
    const float* wt_p    = (const float*)d_in[17];
    float* out = (float*)d_out;

    // ---- workspace layout (256B aligned slabs) ----
    char* ws = (char*)d_ws;
    size_t off = 0;
    auto take = [&](size_t bytes) -> char* {
        char* p = ws + off;
        off += (bytes + 255) & ~(size_t)255;
        return p;
    };
    _Float16* xh    = (_Float16*)take((size_t)NN * FIN * 2);
    _Float16* W1c   = (_Float16*)take(128 * 160 * 2);   // row-major f16
    _Float16* W2p   = (_Float16*)take(128 * CP * 2);
    _Float16* gW2p  = (_Float16*)take(32 * CP * 2);
    _Float16* linWp = (_Float16*)take((size_t)K2P * CP * 2);
    _Float16* W1pk  = (_Float16*)take(128 * 160 * 2);   // fragment-packed
    _Float16* W2pk  = (_Float16*)take(128 * CP * 2);
    _Float16* gW2pk = (_Float16*)take(32 * CP * 2);
    _Float16* linpk = (_Float16*)take((size_t)K2P * CP * 2);
    float*    HG1   = (float*)take((size_t)NN * 160 * 4);
    float*    as1   = (float*)take((size_t)NN * HEADS * 4);
    float*    ad1   = (float*)take((size_t)NN * HEADS * 4);
    float*    nrm   = (float*)take((size_t)NN * 4);
    // ---- zero-initialized region (accumulators, maxes, degrees) ----
    size_t zoff = off;
    float*    deg   = (float*)take((size_t)NN * 4);
    unsigned* m1    = (unsigned*)take((size_t)NN * HEADS * 4);
    float*    s1    = (float*)take((size_t)NN * HEADS * 4);
    float*    out1  = (float*)take((size_t)NN * HH * 4);
    float*    aggG1 = (float*)take((size_t)NN * HID * 4);
    unsigned* m2    = (unsigned*)take((size_t)NN * 4);
    float*    s2    = (float*)take((size_t)NN * 4);
    float*    aggA2 = (float*)take((size_t)NN * CC * 4);
    float*    aggG2 = (float*)take((size_t)NN * CC * 4);
    size_t zwords = (off - zoff) / 4;
    // ---- layer-2 buffers ----
    _Float16* xgatH = (_Float16*)take((size_t)NN * HH * 2);
    _Float16* xgcnH = (_Float16*)take((size_t)NN * HID * 2);
    float*    H2    = (float*)take((size_t)NN * CP * 4);
    float*    G2    = (float*)take((size_t)NN * CP * 4);
    float*    as2   = (float*)take((size_t)NN * 4);
    float*    ad2   = (float*)take((size_t)NN * 4);
    _Float16* catH  = (_Float16*)take((size_t)NN * K2P * 2);
    (void)ws_size; (void)n_in; (void)in_sizes; (void)out_size;

    const int B = 256;
    const int nodeTiles = NN / 16;   // 3125

    // 0) weights -> f16 padded row-major, then fragment-pack; x -> f16; zero
    conv_weights_kernel<<<(128*160 + 128*CP + 32*CP + K2P*CP + B - 1) / B, B, 0, stream>>>(
        gat_W1, gcn_W1, gat_W2, gcn_W2, lin_W, W1c, W2p, gW2p, linWp);
    pack_b_kernel<<<(128 * 160 + B - 1) / B, B, 0, stream>>>(W1c, 160, 128, 10, W1pk);
    pack_b_kernel<<<(128 * CP + B - 1) / B, B, 0, stream>>>(W2p, CP, 128, 3, W2pk);
    pack_b_kernel<<<(32 * CP + B - 1) / B, B, 0, stream>>>(gW2p, CP, 32, 3, gW2pk);
    pack_b_kernel<<<(K2P * CP + B - 1) / B, B, 0, stream>>>(linWp, CP, K2P, 3, linpk);
    f32_to_f16_kernel<<<((size_t)NN * FIN + B - 1) / B, B, 0, stream>>>(x, xh, (size_t)NN * FIN);
    zero_kernel<<<2048, B, 0, stream>>>((float*)(ws + zoff), zwords);

    // 1) GEMM: [H1 | G1] = x @ [gat_W1 | gcn_W1]   (WMMA f16)
    gemm16_kernel<FIN, 128, 10, 160, 160, false><<<nodeTiles, 128, 0, stream>>>(
        xh, W1pk, HG1, nullptr);

    // 2) attention logits + degree norm
    attn1_kernel<<<(NN * HEADS + B - 1) / B, B, 0, stream>>>(HG1, att_s1, att_d1, as1, ad1);
    edge_deg_kernel<<<(ETOT + B - 1) / B, B, 0, stream>>>(ei, deg);
    norm_kernel<<<(NN + B - 1) / B, B, 0, stream>>>(deg, nrm);

    // 3) GAT-1 softmax + fused GAT/GCN aggregation
    edge_max1_kernel<<<(ETOT + B - 1) / B, B, 0, stream>>>(ei, as1, ad1, m1);
    edge_sum1_kernel<<<(ETOT + B - 1) / B, B, 0, stream>>>(ei, as1, ad1, m1, s1);
    edge_agg1_kernel<<<((size_t)ETOT * 32 + B - 1) / B, B, 0, stream>>>(
        ei, as1, ad1, m1, s1, HG1, nrm, out1, aggG1);

    // 4) activations -> f16 inputs of layer 2
    act1_kernel<<<((size_t)NN * 160 + B - 1) / B, B, 0, stream>>>(
        out1, aggG1, gat_b1, gcn_b1, xgatH, xgcnH);

    // 5) layer-2 GEMMs (WMMA)
    gemm16_kernel<HH, 128, 3, CP, CP, false><<<nodeTiles, 128, 0, stream>>>(
        xgatH, W2pk, H2, nullptr);
    gemm16_kernel<HID, 32, 3, CP, CP, false><<<nodeTiles, 128, 0, stream>>>(
        xgcnH, gW2pk, G2, nullptr);
    attn2_kernel<<<(NN + B - 1) / B, B, 0, stream>>>(H2, att_s2, att_d2, as2, ad2);

    // 6) GAT-2 softmax + fused aggregation (incl. GCN-2)
    edge_max2_kernel<<<(ETOT + B - 1) / B, B, 0, stream>>>(ei, as2, ad2, m2);
    edge_sum2_kernel<<<(ETOT + B - 1) / B, B, 0, stream>>>(ei, as2, ad2, m2, s2);
    edge_agg2_kernel<<<((size_t)ETOT * 32 + B - 1) / B, B, 0, stream>>>(
        ei, as2, ad2, m2, s2, H2, G2, nrm, aggA2, aggG2);

    // 7) weighted concat -> f16, final GEMM with bias -> d_out
    cat_kernel<<<((size_t)NN * K2P + B - 1) / B, B, 0, stream>>>(
        aggA2, aggG2, gat_b2, gcn_b2, wc_p, wt_p, catH);
    gemm16_kernel<K2P, 96, 3, CC, CC, true><<<nodeTiles, 128, 0, stream>>>(
        catH, linpk, out, lin_b);
}
